// LabelGCNAttentionRNNv4_25744033972573
// MI455X (gfx1250) — compile-verified, matrix operand order checked by reading, not run on hardware
//
#include <hip/hip_runtime.h>
#include <hip/hip_bf16.h>
#include <math.h>

// ---------- problem constants ----------
static constexpr int BB   = 32;     // batch
static constexpr int SS   = 256;    // seq len
static constexpr int EE   = 300;    // emb dim
static constexpr int EPAD = 320;    // emb dim padded to mult of 32
static constexpr int HH   = 256;    // hidden
static constexpr int LL   = 4000;   // labels
static constexpr int GG   = 512;    // gcn out
static constexpr int L1   = 512;    // mlp hidden 1
static constexpr int L2   = 256;    // mlp hidden 2

typedef __attribute__((ext_vector_type(16))) _Float16 v16h;
typedef __attribute__((ext_vector_type(8)))  _Float16 v8h;
typedef __attribute__((ext_vector_type(8)))  float    v8f;

enum { EP_RELU = 1, EP_TRANS = 2, EP_F16 = 4 };

// ---------- WMMA helpers ----------
__device__ __forceinline__ v8f zero8() {
  v8f z;
#pragma unroll
  for (int i = 0; i < 8; ++i) z[i] = 0.0f;
  return z;
}

// Load a 16x32 f16 operand tile stored row-major [16, ld] (A layout; also used
// for B when B is stored as [N, K] row-major). Matches CDNA5 ISA 16-bit A
// layout: lanes 0-15 hold K {0..7, 16..23}, lanes 16-31 hold K {8..15, 24..31}.
__device__ __forceinline__ v16h frag16(const _Float16* base, int ld) {
  const int lane = threadIdx.x & 31;
  const int row  = lane & 15;
  const int kh   = (lane & 16) ? 8 : 0;
  const _Float16* p = base + row * ld + kh;
  v8h lo = *(const v8h*)(p);
  v8h hi = *(const v8h*)(p + 16);
  v16h f;
#pragma unroll
  for (int i = 0; i < 8; ++i) { f[i] = lo[i]; f[8 + i] = hi[i]; }
  return f;
}

__device__ __forceinline__ v8f wmma32(v16h a, v16h b, v8f c) {
  return __builtin_amdgcn_wmma_f32_16x16x32_f16(false, a, false, b, (short)0, c,
                                                false, false);
}

__device__ __forceinline__ float sigmoidf_(float x) {
  return 1.0f / (1.0f + __expf(-x));
}

// ---------- prep kernels ----------
__global__ void k_embed(const int* __restrict__ tokens,
                        const float* __restrict__ table,
                        _Float16* __restrict__ dst) {
  const int bs = blockIdx.x;              // 0..B*S-1
  const int tok = tokens[bs];
  const int e = threadIdx.x;              // 0..EPAD-1
  dst[(long long)bs * EPAD + e] =
      (e < EE) ? (_Float16)table[(long long)tok * EE + e] : (_Float16)0.0f;
}

__global__ void k_cvt_pad(_Float16* __restrict__ dst,
                          const float* __restrict__ src,
                          int rows, int cols, int dcols) {
  const long long total = (long long)rows * dcols;
  for (long long i = (long long)blockIdx.x * blockDim.x + threadIdx.x; i < total;
       i += (long long)gridDim.x * blockDim.x) {
    const int r = (int)(i / dcols), c = (int)(i % dcols);
    dst[i] = (c < cols) ? (_Float16)src[(long long)r * cols + c]
                        : (_Float16)0.0f;
  }
}

__global__ void k_cvt_T(_Float16* __restrict__ dst,
                        const float* __restrict__ src, int rows, int cols) {
  const long long total = (long long)rows * cols;
  for (long long i = (long long)blockIdx.x * blockDim.x + threadIdx.x; i < total;
       i += (long long)gridDim.x * blockDim.x) {
    const int r = (int)(i / cols), c = (int)(i % cols);
    dst[(long long)c * rows + r] = (_Float16)src[i];
  }
}

// ---------- generic WMMA GEMM: C[M,N] = A[M,K] * B[N,K]^T (+bias,relu) ----------
// block = 256 threads = 8 waves; block tile 32(M) x 64(N); one 16x16 tile/wave.
__global__ __launch_bounds__(256) void k_gemm(
    const _Float16* __restrict__ A, int lda,
    const _Float16* __restrict__ Bm, int ldb,
    float* __restrict__ Cf, _Float16* __restrict__ Ch, int ldc,
    const float* __restrict__ bias, int K, int flags) {
  const int w  = threadIdx.x >> 5;
  const int m0 = blockIdx.y * 32 + (w >> 2) * 16;
  const int n0 = blockIdx.x * 64 + (w & 3) * 16;
  const _Float16* Ap = A  + (long long)m0 * lda;
  const _Float16* Bp = Bm + (long long)n0 * ldb;
  v8f acc = zero8();
  for (int k = 0; k < K; k += 32) {
    __builtin_prefetch(Ap + k + 256, 0, 0);
    __builtin_prefetch(Bp + k + 256, 0, 0);
    v16h a = frag16(Ap + k, lda);
    v16h b = frag16(Bp + k, ldb);
    acc = wmma32(a, b, acc);
  }
  const int lane = threadIdx.x & 31;
  const int col = lane & 15, rb = (lane & 16) ? 8 : 0;
#pragma unroll
  for (int r = 0; r < 8; ++r) {
    const int m = m0 + r + rb, n = n0 + col;
    float v = acc[r];
    if (bias) v += bias[n];
    if (flags & EP_RELU) v = fmaxf(v, 0.0f);
    if (flags & EP_F16) {
      if (flags & EP_TRANS) Ch[(long long)n * ldc + m] = (_Float16)v;
      else                  Ch[(long long)m * ldc + n] = (_Float16)v;
    } else {
      Cf[(long long)m * ldc + n] = v;
    }
  }
}

// ---------- bidirectional LSTM scan ----------
// grid=2 (dir), block=1024 threads = 32 waves. h (f16) and c (f32) resident in
// LDS. Per step: gates[32,1024] = h[32,256] @ whh[1024,256]^T via WMMA; each
// wave owns all four gate tiles (i,f,g,o) of one 16-wide j-slice so the
// nonlinearity epilogue is wave-local. Writes rnn[b,s,:] and rnnT[b,:,s].
__global__ __launch_bounds__(1024) void k_lstm(
    const _Float16* __restrict__ whhF, const _Float16* __restrict__ whhB,
    const float* __restrict__ xgF, const float* __restrict__ xgB,
    _Float16* __restrict__ rnn, _Float16* __restrict__ rnnT) {
  const int dir = blockIdx.x;
  const _Float16* whh = dir ? whhB : whhF;
  const float*    xg  = dir ? xgB  : xgF;
  const int colbase = dir ? HH : 0;

  __shared__ __align__(16) _Float16 hbuf[BB * HH];  // 16 KB
  __shared__ float cbuf[BB * HH];                   // 32 KB
  for (int i = threadIdx.x; i < BB * HH; i += blockDim.x) {
    hbuf[i] = (_Float16)0.0f;
    cbuf[i] = 0.0f;
  }
  __syncthreads();

  const int w = threadIdx.x >> 5;
  const int jt = w & 15, mt = w >> 4;
  const int lane = threadIdx.x & 31;
  const int col = lane & 15, rb = (lane & 16) ? 8 : 0;
  const int j = jt * 16 + col;

  for (int t = 0; t < SS; ++t) {
    const int s = dir ? (SS - 1 - t) : t;
    v8f ai = zero8(), af = zero8(), ag = zero8(), ao = zero8();
    for (int k = 0; k < HH; k += 32) {
      v16h ha = frag16(hbuf + mt * 16 * HH + k, HH);
      v16h bi = frag16(whh + (long long)(0   + jt * 16) * HH + k, HH);
      v16h bf = frag16(whh + (long long)(256 + jt * 16) * HH + k, HH);
      v16h bg = frag16(whh + (long long)(512 + jt * 16) * HH + k, HH);
      v16h bo = frag16(whh + (long long)(768 + jt * 16) * HH + k, HH);
      ai = wmma32(ha, bi, ai);
      af = wmma32(ha, bf, af);
      ag = wmma32(ha, bg, ag);
      ao = wmma32(ha, bo, ao);
    }
    float hreg[8];
#pragma unroll
    for (int r = 0; r < 8; ++r) {
      const int m = mt * 16 + r + rb;
      const long long xi = ((long long)m * SS + s) * 1024;
      const float gi = ai[r] + xg[xi + j];
      const float gf = af[r] + xg[xi + 256 + j];
      const float gg = ag[r] + xg[xi + 512 + j];
      const float go = ao[r] + xg[xi + 768 + j];
      const float iv = sigmoidf_(gi);
      const float fv = sigmoidf_(gf);
      const float gv = tanhf(gg);
      const float ov = sigmoidf_(go);
      const float c  = fv * cbuf[m * HH + j] + iv * gv;  // unique owner per (m,j)
      cbuf[m * HH + j] = c;
      hreg[r] = ov * tanhf(c);
    }
    __syncthreads();  // all waves done reading old h
#pragma unroll
    for (int r = 0; r < 8; ++r) {
      const int m = mt * 16 + r + rb;
      const _Float16 hh = (_Float16)hreg[r];
      hbuf[m * HH + j] = hh;
      rnn [((long long)m * SS + s) * 512 + colbase + j] = hh;
      rnnT[((long long)m * 512 + colbase + j) * SS + s] = hh;
    }
    __syncthreads();  // new h visible before next step
  }
}

// ---------- fused per-(b, 16-label-tile) attention + softmax + MLP ----------
// grid = (L/16, B), block = 256 threads = 8 waves. Shared memory packed into a
// 64 KB union (phase-disjoint reuse) to stay within static-LDS limits.
__global__ __launch_bounds__(256) void k_attn_mlp(
    const int* __restrict__ tokens,
    const _Float16* __restrict__ attnw,   // [L, 512]
    const _Float16* __restrict__ rnn,     // [B, S, 512]
    const _Float16* __restrict__ rnnT,    // [B, 512, S]
    const _Float16* __restrict__ gcn,     // [L, 512] f16
    const _Float16* __restrict__ w1T, const float* __restrict__ b1,
    const _Float16* __restrict__ w2T, const float* __restrict__ b2,
    const float* __restrict__ wout, const float* __restrict__ bout,
    float* __restrict__ out) {
  const int lt = blockIdx.x;  // label tile 0..249
  const int b  = blockIdx.y;  // 0..31

  __shared__ __align__(16) unsigned char smem[65536];
  _Float16* xrow = (_Float16*)(smem);            // [16][1024] phases 3-4
  float*    msk  = (float*)(smem);               // [256]      phases 0-1 (xrow dead)
  float*    sc   = (float*)(smem + 32768);       // [16][256]  phases 1-2
  _Float16* h1s  = (_Float16*)(smem + 32768);    // [16][512]  phases 4-5 (sc dead)
  _Float16* pf   = (_Float16*)(smem + 49152);    // [16][256]  phases 2-3
  float*    h2s  = (float*)(smem + 49152);       // [16][256]  phases 5-6 (pf dead)

  const int tid = threadIdx.x;
  const int w = tid >> 5, lane = tid & 31;
  const int col = lane & 15, rb = (lane & 16) ? 8 : 0;

  // phase 0: additive mask
  if (tid < SS) msk[tid] = (tokens[b * SS + tid] != 0) ? 0.0f : -1.0e9f;
  __syncthreads();

  // phase 1: scores[16, S] = attnw_tile[16,512] @ rnn_b[S,512]^T
  const _Float16* Arow = attnw + (long long)lt * 16 * 512;
  {
    v8f a0 = zero8(), a1 = zero8();
    const _Float16* B0 = rnn + ((long long)b * SS + (2 * w) * 16) * 512;
    const _Float16* B1 = rnn + ((long long)b * SS + (2 * w + 1) * 16) * 512;
    for (int k = 0; k < 512; k += 32) {
      v16h a = frag16(Arow + k, 512);
      a0 = wmma32(a, frag16(B0 + k, 512), a0);
      a1 = wmma32(a, frag16(B1 + k, 512), a1);
    }
#pragma unroll
    for (int r = 0; r < 8; ++r) {
      const int m = r + rb;
      const int s0 = (2 * w) * 16 + col, s1 = (2 * w + 1) * 16 + col;
      sc[m * SS + s0] = a0[r] + msk[s0];
      sc[m * SS + s1] = a1[r] + msk[s1];
    }
  }
  __syncthreads();

  // phase 2: row softmax (wave handles rows 2w, 2w+1)
  for (int rr = 0; rr < 2; ++rr) {
    const int row = 2 * w + rr;
    float mx = -3.0e38f;
    for (int i = lane; i < SS; i += 32) mx = fmaxf(mx, sc[row * SS + i]);
    for (int off = 16; off; off >>= 1) mx = fmaxf(mx, __shfl_xor(mx, off));
    float sum = 0.0f;
    for (int i = lane; i < SS; i += 32) {
      const float e = __expf(sc[row * SS + i] - mx);
      sc[row * SS + i] = e;
      sum += e;
    }
    for (int off = 16; off; off >>= 1) sum += __shfl_xor(sum, off);
    const float inv = 1.0f / sum;
    for (int i = lane; i < SS; i += 32)
      pf[row * SS + i] = (_Float16)(sc[row * SS + i] * inv);
  }
  __syncthreads();

  // phase 3: attn_out[16,512] = P[16,S] @ rnnT_b[512,S]^T  -> xrow[:, 0:512]
  {
    v8f acc[4];
#pragma unroll
    for (int i = 0; i < 4; ++i) acc[i] = zero8();
    for (int k = 0; k < SS; k += 32) {
      v16h a = frag16(pf + k, SS);
#pragma unroll
      for (int i = 0; i < 4; ++i) {
        const int nt = w + 8 * i;
        acc[i] = wmma32(
            a, frag16(rnnT + ((long long)b * 512 + nt * 16) * SS + k, SS),
            acc[i]);
      }
    }
#pragma unroll
    for (int i = 0; i < 4; ++i) {
      const int n0 = (w + 8 * i) * 16;
#pragma unroll
      for (int r = 0; r < 8; ++r)
        xrow[(r + rb) * 1024 + n0 + col] = (_Float16)acc[i][r];
    }
  }
  // phase 3b: append gcn tile -> xrow[:, 512:1024]
  for (int idx = tid; idx < 16 * GG; idx += 256) {
    const int rrow = idx >> 9, cc = idx & 511;
    xrow[rrow * 1024 + 512 + cc] = gcn[((long long)(lt * 16 + rrow)) * GG + cc];
  }
  __syncthreads();

  // phase 4: h1[16,512] = relu(xrow[16,1024] @ w1T[512,1024]^T + b1)
  {
    v8f acc[4];
#pragma unroll
    for (int i = 0; i < 4; ++i) acc[i] = zero8();
    for (int k = 0; k < 1024; k += 32) {
      v16h a = frag16(xrow + k, 1024);
#pragma unroll
      for (int i = 0; i < 4; ++i) {
        const int n0 = (w + 8 * i) * 16;
        acc[i] = wmma32(a, frag16(w1T + (long long)n0 * 1024 + k, 1024), acc[i]);
      }
    }
#pragma unroll
    for (int i = 0; i < 4; ++i) {
      const int n0 = (w + 8 * i) * 16;
#pragma unroll
      for (int r = 0; r < 8; ++r) {
        const int n = n0 + col;
        h1s[(r + rb) * L1 + n] = (_Float16)fmaxf(acc[i][r] + b1[n], 0.0f);
      }
    }
  }
  __syncthreads();

  // phase 5: h2[16,256] = relu(h1[16,512] @ w2T[256,512]^T + b2)
  {
    v8f acc[2];
#pragma unroll
    for (int i = 0; i < 2; ++i) acc[i] = zero8();
    for (int k = 0; k < L1; k += 32) {
      v16h a = frag16(h1s + k, L1);
#pragma unroll
      for (int i = 0; i < 2; ++i) {
        const int n0 = (w + 8 * i) * 16;
        acc[i] = wmma32(a, frag16(w2T + (long long)n0 * L1 + k, L1), acc[i]);
      }
    }
#pragma unroll
    for (int i = 0; i < 2; ++i) {
      const int n0 = (w + 8 * i) * 16;
#pragma unroll
      for (int r = 0; r < 8; ++r) {
        const int n = n0 + col;
        h2s[(r + rb) * L2 + n] = fmaxf(acc[i][r] + b2[n], 0.0f);
      }
    }
  }
  __syncthreads();

  // phase 6: out[b, lt*16+row] = h2[row,:] . w_out + b_out
  for (int rr = 0; rr < 2; ++rr) {
    const int row = 2 * w + rr;
    float p = 0.0f;
    for (int i = lane; i < L2; i += 32) p += h2s[row * L2 + i] * wout[i];
    for (int off = 16; off; off >>= 1) p += __shfl_xor(p, off);
    if (lane == 0) out[(long long)b * LL + lt * 16 + row] = p + bout[0];
  }
}

// ---------- host launch ----------
extern "C" void kernel_launch(void* const* d_in, const int* in_sizes, int n_in,
                              void* d_out, int out_size, void* d_ws,
                              size_t ws_size, hipStream_t stream) {
  (void)in_sizes; (void)n_in; (void)out_size; (void)ws_size;
  const int*   tokens    = (const int*)  d_in[0];
  const float* embtab    = (const float*)d_in[1];
  const float* w_ih_f    = (const float*)d_in[2];
  const float* w_hh_f    = (const float*)d_in[3];
  const float* b_f       = (const float*)d_in[4];
  const float* w_ih_b    = (const float*)d_in[5];
  const float* w_hh_b    = (const float*)d_in[6];
  const float* b_b       = (const float*)d_in[7];
  const float* attn_w    = (const float*)d_in[8];
  const float* label_emb = (const float*)d_in[9];
  const float* adj       = (const float*)d_in[10];
  const float* gcn_w     = (const float*)d_in[11];
  const float* gcn_b     = (const float*)d_in[12];
  const float* w1        = (const float*)d_in[13];
  const float* b1        = (const float*)d_in[14];
  const float* w2        = (const float*)d_in[15];
  const float* b2        = (const float*)d_in[16];
  const float* w_out     = (const float*)d_in[17];
  const float* b_out     = (const float*)d_in[18];
  float* out = (float*)d_out;

  char* ws = (char*)d_ws;
  size_t off = 0;
  auto wsalloc = [&](size_t bytes) -> void* {
    void* p = ws + off;
    off += (bytes + 255) & ~(size_t)255;
    return p;
  };
  _Float16* emb16   = (_Float16*)wsalloc((size_t)BB * SS * EPAD * 2);
  _Float16* wihf16  = (_Float16*)wsalloc((size_t)1024 * EPAD * 2);
  _Float16* wihb16  = (_Float16*)wsalloc((size_t)1024 * EPAD * 2);
  _Float16* whhf16  = (_Float16*)wsalloc((size_t)1024 * HH * 2);
  _Float16* whhb16  = (_Float16*)wsalloc((size_t)1024 * HH * 2);
  float*    xgF     = (float*)   wsalloc((size_t)BB * SS * 1024 * 4);
  float*    xgB     = (float*)   wsalloc((size_t)BB * SS * 1024 * 4);
  _Float16* rnn16   = (_Float16*)wsalloc((size_t)BB * SS * 512 * 2);
  _Float16* rnnT16  = (_Float16*)wsalloc((size_t)BB * 512 * SS * 2);
  _Float16* attnw16 = (_Float16*)wsalloc((size_t)LL * 512 * 2);
  _Float16* lemb16  = (_Float16*)wsalloc((size_t)LL * 256 * 2);
  _Float16* gcnwT16 = (_Float16*)wsalloc((size_t)512 * 256 * 2);
  _Float16* xwT16   = (_Float16*)wsalloc((size_t)512 * LL * 2);
  _Float16* adj16   = (_Float16*)wsalloc((size_t)LL * LL * 2);
  _Float16* gcn16   = (_Float16*)wsalloc((size_t)LL * 512 * 2);
  _Float16* w1T16   = (_Float16*)wsalloc((size_t)512 * 1024 * 2);
  _Float16* w2T16   = (_Float16*)wsalloc((size_t)256 * 512 * 2);

  // --- prep: gather + f32->f16 conversions (with padding / transpose) ---
  k_embed<<<BB * SS, EPAD, 0, stream>>>(tokens, embtab, emb16);
  k_cvt_pad<<<1024, 256, 0, stream>>>(wihf16, w_ih_f, 1024, EE, EPAD);
  k_cvt_pad<<<1024, 256, 0, stream>>>(wihb16, w_ih_b, 1024, EE, EPAD);
  k_cvt_pad<<<1024, 256, 0, stream>>>(whhf16, w_hh_f, 1024, HH, HH);
  k_cvt_pad<<<1024, 256, 0, stream>>>(whhb16, w_hh_b, 1024, HH, HH);
  k_cvt_pad<<<1024, 256, 0, stream>>>(attnw16, attn_w, LL, 512, 512);
  k_cvt_pad<<<1024, 256, 0, stream>>>(lemb16, label_emb, LL, 256, 256);
  k_cvt_pad<<<4096, 256, 0, stream>>>(adj16, adj, LL, LL, LL);
  k_cvt_T<<<1024, 256, 0, stream>>>(gcnwT16, gcn_w, 256, 512);
  k_cvt_T<<<1024, 256, 0, stream>>>(w1T16, w1, 1024, 512);
  k_cvt_T<<<1024, 256, 0, stream>>>(w2T16, w2, 512, 256);

  // --- input-gate projections: xg[B*S,1024] = emb @ w_ih^T + b ---
  {
    dim3 g(1024 / 64, (BB * SS) / 32);  // (16, 256)
    k_gemm<<<g, 256, 0, stream>>>(emb16, EPAD, wihf16, EPAD, xgF, nullptr, 1024,
                                  b_f, EPAD, 0);
    k_gemm<<<g, 256, 0, stream>>>(emb16, EPAD, wihb16, EPAD, xgB, nullptr, 1024,
                                  b_b, EPAD, 0);
  }

  // --- GCN branch: XWt = (label_emb @ gcn_w)^T ; gcn = relu(adj @ XW + b) ---
  {
    dim3 g(GG / 64, LL / 32);  // (8, 125)
    k_gemm<<<g, 256, 0, stream>>>(lemb16, 256, gcnwT16, 256, nullptr, xwT16, LL,
                                  nullptr, 256, EP_TRANS | EP_F16);
    k_gemm<<<g, 256, 0, stream>>>(adj16, LL, xwT16, LL, nullptr, gcn16, GG,
                                  gcn_b, LL, EP_RELU | EP_F16);
  }

  // --- bidirectional LSTM scan (serial over S, LDS-resident state) ---
  k_lstm<<<2, 1024, 0, stream>>>(whhf16, whhb16, xgF, xgB, rnn16, rnnT16);

  // --- fused attention + softmax + concat + MLP -> out[B, L] ---
  {
    dim3 g(LL / 16, BB);  // (250, 32)
    k_attn_mlp<<<g, 256, 0, stream>>>(tokens, attnw16, rnn16, rnnT16, gcn16,
                                      w1T16, b1, w2T16, b2, w_out, b_out, out);
  }
}
